// WFEN_81192061764192
// MI455X (gfx1250) — matrix operand order repaired and checked by reading, not compiled, not run on hardware
//
#include <hip/hip_runtime.h>
#include <hip/hip_bf16.h>
#include <math.h>

typedef __attribute__((ext_vector_type(16))) __bf16 v16bf;
typedef __attribute__((ext_vector_type(8)))  __bf16 v8bf;
typedef __attribute__((ext_vector_type(8)))  float  v8f;

#define HW 16384
#define IMW 128
#define BN 8

// ---- WMMA fragment packing maps (16-bit A / B; contiguous k order, not the ISA table order) ----
// A (MxK tile 16x32): lane<16 holds K 0..15, lane>=16 K 16..31
__device__ __forceinline__ int a_lane(int m, int k) { return m + ((k & 16) ? 16 : 0); }
__device__ __forceinline__ int a_j(int k) { return k & 15; }
// B (KxN tile 32x16): lanes 0-15 hold K 0..15, lanes 16-31 hold K 16..31
__device__ __forceinline__ int b_lane(int n, int k) { return n + ((k & 16) ? 16 : 0); }
__device__ __forceinline__ int b_j(int k) { return k & 15; }

__device__ __forceinline__ v16bf mk16(const __bf16* p) {
  v8bf lo = *(const v8bf*)p;
  v8bf hi = *(const v8bf*)(p + 8);
  return __builtin_shufflevector(lo, hi, 0, 1, 2, 3, 4, 5, 6, 7, 8, 9, 10, 11, 12, 13, 14, 15);
}

// ---------------------------------------------------------------- weight cvt
__global__ void cvt_bf16(const float* __restrict__ s, __bf16* __restrict__ d,
                         int rows, int cs, int cd) {
  int i = blockIdx.x * 256 + threadIdx.x;
  if (i >= rows * cd) return;
  int r = i / cd, c = i % cd;
  d[i] = (c < cs) ? (__bf16)s[r * cs + c] : (__bf16)0.f;
}

// ------------------------------------------- layernorm (+opt add, opt roll)
// dst is PIXEL-MAJOR bf16: [b][pixel][64]
__global__ __launch_bounds__(256) void ln_k(const float* __restrict__ s0,
                                            const float* __restrict__ s1,
                                            const float* __restrict__ w,
                                            const float* __restrict__ bi,
                                            __bf16* __restrict__ dst, int shift) {
  int idx = blockIdx.x * 256 + threadIdx.x;
  int b = idx >> 14, p = idx & (HW - 1);
  const float* p0 = s0 + (size_t)b * 64 * HW + p;
  const float* p1 = s1 ? s1 + (size_t)b * 64 * HW + p : nullptr;
  float xs[64];
  float sum = 0.f, sq = 0.f;
#pragma unroll
  for (int c = 0; c < 64; ++c) {
    float v = p0[(size_t)c * HW];
    if (p1) v += p1[(size_t)c * HW];
    xs[c] = v; sum += v; sq += v * v;
  }
  float mu = sum * (1.f / 64.f);
  float var = sq * (1.f / 64.f) - mu * mu;
  float rs = rsqrtf(var + 1e-5f);
  int hh = p >> 7, ww = p & 127;
  int pr = (((hh - shift) & 127) << 7) | ((ww - shift) & 127);
  __bf16* dp = dst + ((size_t)b * HW + pr) * 64;
#pragma unroll
  for (int g = 0; g < 8; ++g) {
    v8bf t;
#pragma unroll
    for (int j = 0; j < 8; ++j) {
      int c = g * 8 + j;
      t[j] = (__bf16)((xs[c] - mu) * rs * w[c] + bi[c]);
    }
    ((v8bf*)dp)[g] = t;
  }
}

// --------------------------------------------------- conv1x1 GEMM via WMMA
// Y[b,o,p] = sum_c W[o,c] X[b,p,c]  (X pixel-major, Y channel-major) (+res)
template <bool OUTF32, bool ADDRES>
__global__ __launch_bounds__(256) void gemm16(const __bf16* __restrict__ X, int XC,
                                              const __bf16* __restrict__ Wt, int Cin,
                                              int Kpad, int Cout,
                                              const float* __restrict__ res,
                                              void* __restrict__ Y, int YC) {
  __shared__ __bf16 aw[16 * 192];  // A fragments, packed per 32-K chunk
  int tid = threadIdx.x;
  int b = blockIdx.z;
  int o0 = blockIdx.y * 16;
  for (int i = tid; i < 16 * Kpad; i += 256) {
    int r = i / Kpad, k = i % Kpad;
    int o = o0 + r;
    __bf16 g = (o < Cout && k < Cin) ? Wt[o * Cin + k] : (__bf16)0.f;
    int kt = k >> 5, kk = k & 31;
    aw[kt * 512 + a_lane(r, kk) * 16 + a_j(kk)] = g;
  }
  __syncthreads();
  int lane = tid & 31, wave = tid >> 5;
  int p0 = blockIdx.x * 128 + wave * 16;
  int nn = lane & 15;
  int kb = (lane < 16) ? 0 : 16;
  const __bf16* xb = X + ((size_t)b * HW + p0 + nn) * XC + kb;
  v8f acc = {};
  for (int kt = 0; kt < Kpad; kt += 32) {
    v16bf a = mk16(&aw[(kt >> 5) * 512 + lane * 16]);
    v16bf bm = mk16(xb + kt);
    acc = __builtin_amdgcn_wmma_f32_16x16x32_bf16(false, a, false, bm,
                                                  (short)0, acc, false, false);
  }
#pragma unroll
  for (int r = 0; r < 8; ++r) {
    int m = r + ((lane < 16) ? 0 : 8);
    size_t oi = ((size_t)b * YC + o0 + m) * HW + p0 + nn;
    float v = acc[r];
    if (ADDRES) v += res[((size_t)b * 64 + o0 + m) * HW + p0 + nn];
    if (OUTF32) ((float*)Y)[oi] = v;
    else        ((__bf16*)Y)[oi] = (__bf16)v;
  }
}

// ----------------------------------------------------------- depthwise 3x3
// channel-major in/out; each thread computes an 8-pixel strip
__global__ __launch_bounds__(256) void dw3(const __bf16* __restrict__ X,
                                           const __bf16* __restrict__ W9,
                                           __bf16* __restrict__ Y, int C, int Cp) {
  size_t idx = (size_t)blockIdx.x * 256 + threadIdx.x;
  int pw = (int)(idx & 2047);           // strip id: 128 rows x 16 strips
  size_t t = idx >> 11;
  int c = (int)(t % C), b = (int)(t / C);
  int hh = pw >> 4, wb = (pw & 15) * 8;
  const __bf16* xp = X + ((size_t)b * Cp + c) * HW;
  const __bf16* wp = W9 + c * 9;
  float w9[9];
#pragma unroll
  for (int i = 0; i < 9; ++i) w9[i] = (float)wp[i];
  float out[8];
#pragma unroll
  for (int i = 0; i < 8; ++i) out[i] = 0.f;
#pragma unroll
  for (int r = 0; r < 3; ++r) {
    int h2 = hh + r - 1;
    if (h2 < 0 || h2 >= IMW) continue;
    const __bf16* row = xp + h2 * IMW;
    float seg[10];
#pragma unroll
    for (int j = 0; j < 10; ++j) {
      int col = wb - 1 + j;
      seg[j] = (col >= 0 && col < IMW) ? (float)row[col] : 0.f;
    }
#pragma unroll
    for (int dx = 0; dx < 3; ++dx)
#pragma unroll
      for (int i = 0; i < 8; ++i) out[i] += seg[i + dx] * w9[r * 3 + dx];
  }
  v8bf ov;
#pragma unroll
  for (int i = 0; i < 8; ++i) ov[i] = (__bf16)out[i];
  *(v8bf*)&Y[((size_t)b * Cp + c) * HW + hh * IMW + wb] = ov;
}

// --------------------------------------------------------- gated-GeLU (FFN)
// in: channel-major [b][352][HW]; out: pixel-major [b][HW][192] (pad zeroed)
__global__ __launch_bounds__(256) void gate_k(const __bf16* __restrict__ H,
                                              __bf16* __restrict__ G) {
  int p = blockIdx.x * 256 + threadIdx.x;
  int c8 = blockIdx.y * 8;
  int b = blockIdx.z;
  v8bf ov;
#pragma unroll
  for (int j = 0; j < 8; ++j) {
    int c = c8 + j;
    float g = 0.f;
    if (c < 170) {
      float x1 = (float)H[((size_t)b * 352 + c) * HW + p];
      float x2 = (float)H[((size_t)b * 352 + 170 + c) * HW + p];
      g = 0.5f * x1 * (1.f + erff(x1 * 0.70710678118f)) * x2;
    }
    ov[j] = (__bf16)g;
  }
  *(v8bf*)&G[((size_t)b * HW + p) * 192 + c8] = ov;
}

// --------------------------------------------- RSA: windowed channel attn
// one block per 8x8 window; fragment-packed LDS; output pixel-major bf16
__global__ __launch_bounds__(256) void rsa_attn(const __bf16* __restrict__ QKV,
                                                const float* __restrict__ tptr,
                                                __bf16* __restrict__ O) {
  __shared__ __bf16 qraw[4096], kraw[4096];
  __shared__ __bf16 qA[4096], kB[4096], vA[4096], atB[4096];
  __shared__ float nrm[128];
  int tid = threadIdx.x;
  int b = blockIdx.x >> 8, wi = blockIdx.x & 255;
  int h0 = (wi >> 4) * 8, w0 = (wi & 15) * 8;
  const size_t bb = (size_t)b * 192 * HW;
  for (int t = tid; t < 3 * 4096; t += 256) {
    int which = t >> 12, r = t & 4095;
    int c = r >> 6, pos = r & 63;
    int py = pos >> 3, px = pos & 7;
    __bf16 g = QKV[bb + (size_t)(which * 64 + c) * HW + (h0 + py) * IMW + (w0 + px)];
    if (which == 0) qraw[pos * 64 + c] = g;
    else if (which == 1) kraw[pos * 64 + c] = g;
    else {  // v: A-pack  (M=pos, K=c)
      int pt = pos >> 4, m = pos & 15, cch = c >> 5, kk = c & 31;
      vA[(pt * 2 + cch) * 512 + a_lane(m, kk) * 16 + a_j(kk)] = g;
    }
  }
  __syncthreads();
  if (tid < 128) {  // l2norm scale per position, q then k
    int which = tid >> 6, pos = tid & 63;
    const __bf16* base = (which ? kraw : qraw) + pos * 64;
    float ss = 0.f;
#pragma unroll
    for (int g = 0; g < 8; ++g) {
      v8bf v = *(const v8bf*)(base + g * 8);
#pragma unroll
      for (int j = 0; j < 8; ++j) { float f = (float)v[j]; ss += f * f; }
    }
    nrm[tid] = 1.f / fmaxf(sqrtf(ss), 1e-12f);
  }
  __syncthreads();
  for (int t = tid; t < 2 * 4096; t += 256) {  // scale + fragment-pack q,k
    int which = t >> 12, r = t & 4095;
    int c = r >> 6, pos = r & 63;
    float sc = nrm[which * 64 + pos];
    __bf16 g = (__bf16)((float)((which ? kraw : qraw)[pos * 64 + c]) * sc);
    if (which == 0) {  // q: A-pack (M=c, K=pos)
      int ct = c >> 4, m = c & 15, pch = pos >> 5, kk = pos & 31;
      qA[(ct * 2 + pch) * 512 + a_lane(m, kk) * 16 + a_j(kk)] = g;
    } else {           // k: B-pack (K=pos, N=c)
      int dt = c >> 4, n = c & 15, pch = pos >> 5, kk = pos & 31;
      kB[(pch * 4 + dt) * 512 + b_lane(n, kk) * 16 + b_j(kk)] = g;
    }
  }
  __syncthreads();
  float temp = tptr[0];
  int lane = tid & 31, wave = tid >> 5;
  int nn = lane & 15;
  // GEMM1: attn[c,d] = sum_p q[p,c] k[p,d] -> relu(*temp) -> B-pack in atB
  for (int ti = wave * 2; ti < wave * 2 + 2; ++ti) {
    int ct = ti >> 2, dt = ti & 3;
    v8f acc = {};
#pragma unroll
    for (int pch = 0; pch < 2; ++pch) {
      v16bf a = mk16(&qA[(ct * 2 + pch) * 512 + lane * 16]);
      v16bf bm = mk16(&kB[(pch * 4 + dt) * 512 + lane * 16]);
      acc = __builtin_amdgcn_wmma_f32_16x16x32_bf16(false, a, false, bm,
                                                    (short)0, acc, false, false);
    }
#pragma unroll
    for (int r = 0; r < 8; ++r) {
      int m = r + ((lane < 16) ? 0 : 8);
      int c = ct * 16 + m;
      int cch = c >> 5, kk = c & 31;
      atB[(cch * 4 + dt) * 512 + b_lane(nn, kk) * 16 + b_j(kk)] =
          (__bf16)fmaxf(acc[r] * temp, 0.f);
    }
  }
  __syncthreads();
  // GEMM2: out[p,d] = sum_c v[p,c] attn[c,d]; store rolled back (+4,+4)
  for (int ti = wave * 2; ti < wave * 2 + 2; ++ti) {
    int pt = ti >> 2, dt = ti & 3;
    int d0 = dt * 16;
    v8f acc = {};
#pragma unroll
    for (int cch = 0; cch < 2; ++cch) {
      v16bf a = mk16(&vA[(pt * 2 + cch) * 512 + lane * 16]);
      v16bf bm = mk16(&atB[(cch * 4 + dt) * 512 + lane * 16]);
      acc = __builtin_amdgcn_wmma_f32_16x16x32_bf16(false, a, false, bm,
                                                    (short)0, acc, false, false);
    }
#pragma unroll
    for (int r = 0; r < 8; ++r) {
      int m = r + ((lane < 16) ? 0 : 8);
      int pos = pt * 16 + m, py = pos >> 3, px = pos & 7;
      int hh = (h0 + py + 4) & 127, ww = (w0 + px + 4) & 127;
      O[((size_t)b * HW + hh * IMW + ww) * 64 + d0 + nn] = (__bf16)acc[r];
    }
  }
}

// --------------------------------------------- GSA: global channel attention
// one block per (batch, head); output pixel-major bf16
__global__ __launch_bounds__(256) void gsa_k(const __bf16* __restrict__ QKV,
                                             const float* __restrict__ tptr,
                                             __bf16* __restrict__ O) {
  int b = blockIdx.x >> 3, hd = blockIdx.x & 7;
  int tid = threadIdx.x;
  const size_t base = (size_t)b * 192 * HW;
  const __bf16* qp = QKV + base + (size_t)(hd * 8) * HW;
  const __bf16* kp = QKV + base + (size_t)(64 + hd * 8) * HW;
  const __bf16* vp = QKV + base + (size_t)(128 + hd * 8) * HW;
  float G[80];
#pragma unroll
  for (int i = 0; i < 80; ++i) G[i] = 0.f;
  for (int n = tid; n < HW; n += 256) {
    float qv[8], kv[8];
#pragma unroll
    for (int c = 0; c < 8; ++c) {
      qv[c] = (float)qp[(size_t)c * HW + n];
      kv[c] = (float)kp[(size_t)c * HW + n];
    }
#pragma unroll
    for (int c = 0; c < 8; ++c) {
      G[64 + c] += qv[c] * qv[c];
      G[72 + c] += kv[c] * kv[c];
#pragma unroll
      for (int d = 0; d < 8; ++d) G[c * 8 + d] += qv[c] * kv[d];
    }
  }
  int lane = tid & 31, wave = tid >> 5;
#pragma unroll
  for (int i = 0; i < 80; ++i)
    for (int off = 16; off > 0; off >>= 1) G[i] += __shfl_xor(G[i], off, 32);
  __shared__ float red[8][80];
  if (lane == 0)
#pragma unroll
    for (int i = 0; i < 80; ++i) red[wave][i] = G[i];
  __syncthreads();
  __shared__ float fin[80];
  if (tid < 80) {
    float s = 0.f;
    for (int w = 0; w < 8; ++w) s += red[w][tid];
    fin[tid] = s;
  }
  __syncthreads();
  __shared__ float attn[64];
  if (tid < 64) {
    int c = tid >> 3, d = tid & 7;
    float s = fin[tid] / (fmaxf(sqrtf(fin[64 + c]), 1e-12f) *
                          fmaxf(sqrtf(fin[72 + d]), 1e-12f));
    attn[tid] = fmaxf(s * tptr[0], 0.f);
  }
  __syncthreads();
  for (int n = tid; n < HW; n += 256) {
    float vv[8];
#pragma unroll
    for (int d = 0; d < 8; ++d) vv[d] = (float)vp[(size_t)d * HW + n];
#pragma unroll
    for (int c = 0; c < 8; ++c) {
      float s = 0.f;
#pragma unroll
      for (int d = 0; d < 8; ++d) s += attn[c * 8 + d] * vv[d];
      O[((size_t)b * HW + n) * 64 + c * 8 + hd] = (__bf16)s;  // ch = cp*HEADS+head
    }
  }
}

// ---------------------------------------------------------------- launcher
extern "C" void kernel_launch(void* const* d_in, const int* in_sizes, int n_in,
                              void* d_out, int out_size, void* d_ws, size_t ws_size,
                              hipStream_t stream) {
  (void)in_sizes; (void)n_in; (void)out_size; (void)ws_size;
  const float* x        = (const float*)d_in[0];
  const float* ln_s0_w  = (const float*)d_in[1];
  const float* ln_s0_b  = (const float*)d_in[2];
  const float* ln_s2_w  = (const float*)d_in[3];
  const float* ln_s2_b  = (const float*)d_in[4];
  const float* rsa_qkv  = (const float*)d_in[5];
  const float* rsa_dw   = (const float*)d_in[6];
  const float* rsa_proj = (const float*)d_in[7];
  const float* rsa_temp = (const float*)d_in[8];
  const float* ffs_in   = (const float*)d_in[9];
  const float* ffs_dw   = (const float*)d_in[10];
  const float* ffs_out  = (const float*)d_in[11];
  const float* ln_c0_w  = (const float*)d_in[12];
  const float* ln_c0_b  = (const float*)d_in[13];
  const float* ln_c2_w  = (const float*)d_in[14];
  const float* ln_c2_b  = (const float*)d_in[15];
  const float* gsa_qkv  = (const float*)d_in[16];
  const float* gsa_dw   = (const float*)d_in[17];
  const float* gsa_proj = (const float*)d_in[18];
  const float* gsa_temp = (const float*)d_in[19];
  const float* ffc_in   = (const float*)d_in[20];
  const float* ffc_dw   = (const float*)d_in[21];
  const float* ffc_out  = (const float*)d_in[22];

  char* ws = (char*)d_ws;
  const size_t szF = (size_t)BN * 64 * HW * 4;    // f32 state (channel-major)
  const size_t szL = (size_t)BN * 64 * HW * 2;    // bf16 64ch  (pixel-major)
  const size_t szT = (size_t)BN * 352 * HW * 2;   // bf16 wide  (channel-major)
  float*  A   = (float*)ws;                       // running state
  float*  Bf  = (float*)(ws + szF);               // module output x_
  __bf16* L16 = (__bf16*)(ws + 2 * szF);          // pixel-major GEMM input
  __bf16* Ta  = (__bf16*)(ws + 2 * szF + szL);    // wide ch-major / gate out p-major
  __bf16* Tb  = (__bf16*)(ws + 2 * szF + szL + szT);
  __bf16* Wb  = (__bf16*)(ws + 2 * szF + szL + 2 * szT);

  const int wo_rq = 0,      wo_rd = 12288,  wo_rp = 14016, wo_fi = 18112;
  const int wo_fd = 39872,  wo_fo = 42932,  wo_gq = 55220, wo_gd = 67508;
  const int wo_gp = 69236,  wo_ci = 73332,  wo_cd = 95092, wo_co = 98152;

  auto cvt = [&](const float* s, int off, int r, int cs, int cd) {
    int tot = r * cd;
    cvt_bf16<<<(tot + 255) / 256, 256, 0, stream>>>(s, Wb + off, r, cs, cd);
  };
  cvt(rsa_qkv,  wo_rq, 192, 64, 64);
  cvt(rsa_dw,   wo_rd, 192, 9, 9);
  cvt(rsa_proj, wo_rp, 64, 64, 64);
  cvt(ffs_in,   wo_fi, 340, 64, 64);
  cvt(ffs_dw,   wo_fd, 340, 9, 9);
  cvt(ffs_out,  wo_fo, 64, 170, 192);  // K zero-padded to 192
  cvt(gsa_qkv,  wo_gq, 192, 64, 64);
  cvt(gsa_dw,   wo_gd, 192, 9, 9);
  cvt(gsa_proj, wo_gp, 64, 64, 64);
  cvt(ffc_in,   wo_ci, 340, 64, 64);
  cvt(ffc_dw,   wo_cd, 340, 9, 9);
  cvt(ffc_out,  wo_co, 64, 170, 192);

  dim3 blk(256);
  // ---- RSA half ----
  ln_k<<<512, blk, 0, stream>>>(x, nullptr, ln_s0_w, ln_s0_b, L16, 4);  // roll(-4)
  gemm16<false, false><<<dim3(128, 12, 8), blk, 0, stream>>>(L16, 64, Wb + wo_rq, 64, 64, 192, nullptr, Ta, 192);
  dw3<<<12288, blk, 0, stream>>>(Ta, Wb + wo_rd, Tb, 192, 192);
  rsa_attn<<<2048, blk, 0, stream>>>(Tb, rsa_temp, L16);
  gemm16<true, false><<<dim3(128, 4, 8), blk, 0, stream>>>(L16, 64, Wb + wo_rp, 64, 64, 64, nullptr, Bf, 64);  // x_ = Bf
  ln_k<<<512, blk, 0, stream>>>(Bf, x, ln_s2_w, ln_s2_b, L16, 0);       // LN(x_+x)
  gemm16<false, false><<<dim3(128, 22, 8), blk, 0, stream>>>(L16, 64, Wb + wo_fi, 64, 64, 340, nullptr, Ta, 352);
  dw3<<<21760, blk, 0, stream>>>(Ta, Wb + wo_fd, Tb, 340, 352);
  gate_k<<<dim3(64, 24, 8), blk, 0, stream>>>(Tb, Ta);
  gemm16<true, true><<<dim3(128, 4, 8), blk, 0, stream>>>(Ta, 192, Wb + wo_fo, 192, 192, 64, Bf, A, 64);  // x = ffn+x_

  // ---- GSA half ----
  ln_k<<<512, blk, 0, stream>>>(A, nullptr, ln_c0_w, ln_c0_b, L16, 0);
  gemm16<false, false><<<dim3(128, 12, 8), blk, 0, stream>>>(L16, 64, Wb + wo_gq, 64, 64, 192, nullptr, Ta, 192);
  dw3<<<12288, blk, 0, stream>>>(Ta, Wb + wo_gd, Tb, 192, 192);
  gsa_k<<<64, blk, 0, stream>>>(Tb, gsa_temp, L16);
  gemm16<true, false><<<dim3(128, 4, 8), blk, 0, stream>>>(L16, 64, Wb + wo_gp, 64, 64, 64, nullptr, Bf, 64);  // x_ = Bf
  ln_k<<<512, blk, 0, stream>>>(Bf, A, ln_c2_w, ln_c2_b, L16, 0);
  gemm16<false, false><<<dim3(128, 22, 8), blk, 0, stream>>>(L16, 64, Wb + wo_ci, 64, 64, 340, nullptr, Ta, 352);
  dw3<<<21760, blk, 0, stream>>>(Ta, Wb + wo_cd, Tb, 340, 352);
  gate_k<<<dim3(64, 24, 8), blk, 0, stream>>>(Tb, Ta);
  gemm16<true, true><<<dim3(128, 4, 8), blk, 0, stream>>>(Ta, 192, Wb + wo_co, 192, 192, 64, Bf, (float*)d_out, 64);
}
